// GCN_67577015436437
// MI455X (gfx1250) — matrix operand order repaired and checked by reading, not compiled, not run on hardware
//
#include <hip/hip_runtime.h>
#include <hip/hip_bf16.h>
#include <stdint.h>

// ---------------------------------------------------------------------------
// Graph-WaveNet GCN for MI455X (gfx1250), bf16 WMMA with fp32 accumulation.
//   Phase P: At[2s]  = a_s^T  (bf16),  At[2s+1] = (a_s*a_s)^T (fp32 acc -> bf16)
//   Phase X: Xb[m][v] = x[n][c][v][l], m=(n*C+c)*L+l  (bf16, transposed)
//   Phase G: Y_j[m][w] = sum_v Xb[m][v] * At_j[w][v]   (wmma, async-LDS pipeline)
//   Phase O: out[n][o][w][l] = b[o] + sum_cc Wb[o][cc]*H[cc][(n,w,l)]  (wmma)
// ---------------------------------------------------------------------------

typedef __attribute__((ext_vector_type(16))) __bf16 v16bf;
typedef __attribute__((ext_vector_type(8)))  __bf16 v8bf;
typedef __attribute__((ext_vector_type(8)))  float  v8f;

#define B_  64
#define C_  32
#define V_  512
#define L_  13
#define CO_ 64
#define M_  (B_*C_*L_)   // 26624
#define CC_ 224          // 7 * 32 concat channels
#define CCP 240          // padded H-tile row stride (480 B: 32B-aligned, banks spread)
#define AS_ 48           // padded A/B tile row stride in bf16 (96 B)

#ifndef __has_builtin
#define __has_builtin(x) 0
#endif

#if __has_builtin(__builtin_amdgcn_global_load_async_to_lds_b128)
#define USE_ASYNC 1
#else
#define USE_ASYNC 0
#endif

#if USE_ASYNC
// Builtin prototype (from hipcc diagnostic): param0 = GCC-vector int4 in AS1
// (global), param1 = int4 in AS3 (LDS), then immediate offset and cpol.
typedef int v4i_ __attribute__((vector_size(16)));
typedef __attribute__((address_space(1))) v4i_* as1_v4i;
typedef __attribute__((address_space(3))) v4i_* as3_v4i;

static __device__ __forceinline__ void async_cp16(void* lds_dst, const void* gsrc) {
  __builtin_amdgcn_global_load_async_to_lds_b128((as1_v4i)gsrc, (as3_v4i)lds_dst,
                                                 /*offset=*/0, /*cpol=*/0);
}
template <int N> static __device__ __forceinline__ void wait_async() {
#if __has_builtin(__builtin_amdgcn_s_wait_asynccnt)
  __builtin_amdgcn_s_wait_asynccnt(N);
#else
  asm volatile("s_wait_asynccnt %0" :: "i"(N) : "memory");
#endif
}
#endif

static __device__ __forceinline__ __bf16 f2bf(float f) {
  unsigned u = __builtin_bit_cast(unsigned, f);
  u += 0x7FFFu + ((u >> 16) & 1u);                // round-to-nearest-even
  unsigned short h = (unsigned short)(u >> 16);
  return __builtin_bit_cast(__bf16, h);
}

static __device__ __forceinline__ v8f vzero8() {
  v8f z;
#pragma unroll
  for (int i = 0; i < 8; ++i) z[i] = 0.0f;
  return z;
}

union FragU { v16bf v; v8bf h[2]; };

// ---------------------------------------------------------------------------
// Phase P: support prep.  grid (32 vTiles, 32 wTiles, 3 supports), block 16x16
// ---------------------------------------------------------------------------
__global__ void __launch_bounds__(256)
prep_supports(const float* __restrict__ s0, const float* __restrict__ s1,
              const float* __restrict__ s2, __bf16* __restrict__ At) {
  const float* a = (blockIdx.z == 0) ? s0 : (blockIdx.z == 1 ? s1 : s2);
  __bf16* At1 = At + (size_t)(2 * blockIdx.z) * V_ * V_;
  __bf16* At2 = At1 + (size_t)V_ * V_;
  const int v0 = blockIdx.x * 16, w0 = blockIdx.y * 16;
  const int tx = threadIdx.x, ty = threadIdx.y;      // tx -> v, ty -> w
  const int tid = ty * 16 + tx;

  __shared__ float Lt[16][64];   // a[v0+vv][u0+uu]
  __shared__ float Rt[64][17];   // a[u0+uu][w0+ww] (+1 pad vs bank conflicts)

  float acc = 0.0f;
  for (int u0 = 0; u0 < V_; u0 += 64) {
#pragma unroll
    for (int i = 0; i < 4; ++i) {
      int e = tid + i * 256;
      Lt[e >> 6][e & 63] = a[(size_t)(v0 + (e >> 6)) * V_ + u0 + (e & 63)];
      Rt[e >> 4][e & 15] = a[(size_t)(u0 + (e >> 4)) * V_ + w0 + (e & 15)];
    }
    __syncthreads();
#pragma unroll
    for (int u = 0; u < 64; ++u) acc += Lt[tx][u] * Rt[u][ty];
    __syncthreads();
  }
  At2[(size_t)(w0 + ty) * V_ + v0 + tx] = f2bf(acc);
  At1[(size_t)(w0 + ty) * V_ + v0 + tx] = f2bf(a[(size_t)(v0 + tx) * V_ + (w0 + ty)]);
}

// ---------------------------------------------------------------------------
// Phase X: x -> Xb[m][v] bf16 (per-(n,c) 512x13 -> 13x512 transpose via LDS)
// ---------------------------------------------------------------------------
__global__ void __launch_bounds__(256)
convert_x(const float* __restrict__ x, __bf16* __restrict__ Xb) {
  __shared__ __bf16 lds[V_ * L_];                 // 13 KB
  const int nc = blockIdx.x;
  const float* src = x + (size_t)nc * V_ * L_;
  for (int e = threadIdx.x; e < V_ * L_; e += 256) lds[e] = f2bf(src[e]);
  __syncthreads();
  __bf16* dst = Xb + (size_t)nc * L_ * V_;        // rows m = nc*L + l
  for (int e = threadIdx.x; e < V_ * L_; e += 256) {
    int l = e >> 9, v = e & 511;                  // 6656 = 13 * 512
    dst[(size_t)l * V_ + v] = lds[v * L_ + l];
  }
}

__global__ void convert_W(const float* __restrict__ W, __bf16* __restrict__ Wb) {
  int e = blockIdx.x * blockDim.x + threadIdx.x;
  if (e < CO_ * CC_) Wb[e] = f2bf(W[e]);
}

// ---------------------------------------------------------------------------
// Phase G: Y_j[m][w] = sum_v Xb[m][v] * At_j[w][v]
// grid (208 mBlk, 4 wBlk, 6 j), block 256 (8 waves), tile 128m x 128w, K-step 32
// Double-buffered LDS; async global->LDS copies when the toolchain exposes them.
// ---------------------------------------------------------------------------
__global__ void __launch_bounds__(256)
prop_gemm(const __bf16* __restrict__ Xb, const __bf16* __restrict__ At,
          __bf16* __restrict__ Yb) {
  const int m0 = blockIdx.x * 128;
  const int w0 = blockIdx.y * 128;
  const int j  = blockIdx.z;
  const __bf16* Aj = At + (size_t)j * V_ * V_;
  __bf16* Yj = Yb + (size_t)j * M_ * V_;

  __shared__ __align__(32) __bf16 ldsA[2][128 * AS_];  // 12 KB each, [m][v] rows
  __shared__ __align__(32) __bf16 ldsB[2][128 * AS_];  // 12 KB each, [w][v] rows

  const int tid  = threadIdx.x;
  const int lane = tid & 31, wid = tid >> 5;      // wave32
  const int waveM = wid & 3, waveN = wid >> 2;    // 4 x 2 wave grid
  const int l16 = lane & 15, lh = lane >> 4;

  // staging coords: 512 16B-chunks per tile, 2 per thread per tile
  const int r0 = tid >> 2,           c0 = (tid & 3) * 8;
  const int r1 = (tid + 256) >> 2,   c1 = ((tid + 256) & 3) * 8;

  auto stage = [&](int buf, int v0) {
#if USE_ASYNC
    async_cp16(&ldsA[buf][r0 * AS_ + c0], &Xb[(size_t)(m0 + r0) * V_ + v0 + c0]);
    async_cp16(&ldsA[buf][r1 * AS_ + c1], &Xb[(size_t)(m0 + r1) * V_ + v0 + c1]);
    async_cp16(&ldsB[buf][r0 * AS_ + c0], &Aj[(size_t)(w0 + r0) * V_ + v0 + c0]);
    async_cp16(&ldsB[buf][r1 * AS_ + c1], &Aj[(size_t)(w0 + r1) * V_ + v0 + c1]);
#else
    *(uint4*)&ldsA[buf][r0 * AS_ + c0] = *(const uint4*)&Xb[(size_t)(m0 + r0) * V_ + v0 + c0];
    *(uint4*)&ldsA[buf][r1 * AS_ + c1] = *(const uint4*)&Xb[(size_t)(m0 + r1) * V_ + v0 + c1];
    *(uint4*)&ldsB[buf][r0 * AS_ + c0] = *(const uint4*)&Aj[(size_t)(w0 + r0) * V_ + v0 + c0];
    *(uint4*)&ldsB[buf][r1 * AS_ + c1] = *(const uint4*)&Aj[(size_t)(w0 + r1) * V_ + v0 + c1];
#endif
  };

  v8f acc[2][4];
#pragma unroll
  for (int ti = 0; ti < 2; ++ti)
#pragma unroll
    for (int tj = 0; tj < 4; ++tj) acc[ti][tj] = vzero8();

  stage(0, 0);
  for (int k = 0; k < V_ / 32; ++k) {
    const int cur = k & 1;
    if (k + 1 < V_ / 32) stage(cur ^ 1, (k + 1) * 32);   // overlap next K-step
#if USE_ASYNC
    if (k + 1 < V_ / 32) wait_async<4>(); else wait_async<0>();
#endif
    __syncthreads();                                     // cur tiles visible

    v16bf afrag[2], bfrag[4];
#pragma unroll
    for (int ti = 0; ti < 2; ++ti) {     // A: lane row M, K runs {k0..k0+7, k0+16..k0+23}
      int arow = waveM * 32 + ti * 16 + l16;
      FragU t;
      t.h[0] = *(const v8bf*)&ldsA[cur][arow * AS_ + lh * 8];
      t.h[1] = *(const v8bf*)&ldsA[cur][arow * AS_ + lh * 8 + 16];
      afrag[ti] = t.v;
    }
#pragma unroll
    for (int tj = 0; tj < 4; ++tj) {     // B: lane column N, K runs {k0..k0+7, k0+16..k0+23}
      int brow = waveN * 64 + tj * 16 + l16;
      FragU u;
      u.h[0] = *(const v8bf*)&ldsB[cur][brow * AS_ + lh * 8];
      u.h[1] = *(const v8bf*)&ldsB[cur][brow * AS_ + lh * 8 + 16];
      bfrag[tj] = u.v;
    }
#pragma unroll
    for (int ti = 0; ti < 2; ++ti)
#pragma unroll
      for (int tj = 0; tj < 4; ++tj)
        acc[ti][tj] = __builtin_amdgcn_wmma_f32_16x16x32_bf16(
            false, afrag[ti], false, bfrag[tj], (short)0, acc[ti][tj],
            false, false);
    __syncthreads();                     // all reads of `cur` done before reuse
  }

#pragma unroll
  for (int ti = 0; ti < 2; ++ti)
#pragma unroll
    for (int tj = 0; tj < 4; ++tj)
#pragma unroll
      for (int r = 0; r < 8; ++r) {
        int m = m0 + waveM * 32 + ti * 16 + r + lh * 8;
        int w = w0 + waveN * 64 + tj * 16 + l16;
        Yj[(size_t)m * V_ + w] = f2bf(acc[ti][tj][r]);
      }
}

// ---------------------------------------------------------------------------
// Phase O: out[n][o][w][l] = b[o] + sum_cc Wb[o][cc] * H[cc][(n,w,l)]
// grid (64 wChunks of 8, 64 n), block 256. LDS H tile [112 p'][240 cc] bf16.
// ---------------------------------------------------------------------------
__global__ void __launch_bounds__(256)
out_gemm(const float* __restrict__ x, const __bf16* __restrict__ Yb,
         const __bf16* __restrict__ Wb, const float* __restrict__ bias,
         float* __restrict__ out) {
  const int w0 = blockIdx.x * 8;
  const int n  = blockIdx.y;
  __shared__ __align__(32) __bf16 H[112 * CCP];   // 52.5 KB, p' = w'*13 + l

  const int tid = threadIdx.x, lane = tid & 31, wid = tid >> 5;
  const int l16 = lane & 15, lh = lane >> 4;

  // stage concat block 0: x[n][c][w0+w'][l] -> H[p'][c]
  {
    int c = tid >> 3, wp = tid & 7;               // 256 = 32 * 8
    const float* src = x + (((size_t)(n * C_ + c) * V_) + (w0 + wp)) * L_;
#pragma unroll
    for (int l = 0; l < L_; ++l)
      H[(wp * L_ + l) * CCP + c] = f2bf(src[l]);
  }
  // stage blocks 1..6: Y_j rows -> H[p'][32 + j*32 + c]
  for (int rid = tid; rid < 6 * C_ * L_; rid += 256) {
    int jj  = rid / (C_ * L_);
    int rem = rid - jj * (C_ * L_);
    int c = rem / L_, l = rem - c * L_;
    const __bf16* src = Yb + (size_t)jj * M_ * V_ +
                        (size_t)((n * C_ + c) * L_ + l) * V_ + w0;
    v8bf vals = *(const v8bf*)src;                // 8 consecutive w
#pragma unroll
    for (int wp = 0; wp < 8; ++wp)
      H[(wp * L_ + l) * CCP + 32 + jj * C_ + c] = vals[wp];
  }
  __syncthreads();

  if (wid < 7) {
    const int pt = wid;                           // one 16-wide p' tile per wave
    v8f acc[4];
#pragma unroll
    for (int ot = 0; ot < 4; ++ot) acc[ot] = vzero8();

    for (int ks = 0; ks < 7; ++ks) {
      int cc0 = ks * 32;
      FragU bu;
      bu.h[0] = *(const v8bf*)&H[(pt * 16 + l16) * CCP + cc0 + lh * 8];
      bu.h[1] = *(const v8bf*)&H[(pt * 16 + l16) * CCP + cc0 + lh * 8 + 16];
      v16bf bfrag = bu.v;
#pragma unroll
      for (int ot = 0; ot < 4; ++ot) {
        int o = ot * 16 + l16;
        FragU t;                                   // A frag from hot Wb (28 KB)
        t.h[0] = *(const v8bf*)&Wb[o * CC_ + cc0 + lh * 8];
        t.h[1] = *(const v8bf*)&Wb[o * CC_ + cc0 + lh * 8 + 16];
        acc[ot] = __builtin_amdgcn_wmma_f32_16x16x32_bf16(
            false, t.v, false, bfrag, (short)0, acc[ot], false, false);
      }
    }

    int p = pt * 16 + l16;
    if (p < 104) {                                // skip pad columns
      int wp = p / L_, l = p - wp * L_;
#pragma unroll
      for (int ot = 0; ot < 4; ++ot)
#pragma unroll
        for (int r = 0; r < 8; ++r) {
          int o = ot * 16 + r + lh * 8;
          out[(((size_t)(n * CO_ + o)) * V_ + (w0 + wp)) * L_ + l] =
              acc[ot][r] + bias[o];
        }
    }
  }
}

// ---------------------------------------------------------------------------
extern "C" void kernel_launch(void* const* d_in, const int* in_sizes, int n_in,
                              void* d_out, int out_size, void* d_ws, size_t ws_size,
                              hipStream_t stream) {
  const float* x  = (const float*)d_in[0];
  const float* s0 = (const float*)d_in[1];
  const float* s1 = (const float*)d_in[2];
  const float* s2 = (const float*)d_in[3];
  const float* W  = (const float*)d_in[4];
  const float* b  = (const float*)d_in[5];
  float* out = (float*)d_out;

  // workspace layout (bytes), all 256B aligned; total ~194 MB
  char* ws = (char*)d_ws;
  __bf16* Xb = (__bf16*)(ws);                 // 26624*512*2   = 27,262,976
  __bf16* At = (__bf16*)(ws + 27262976);      // 6*512*512*2   =  3,145,728
  __bf16* Wb = (__bf16*)(ws + 30408704);      // 64*224*2 (pad to 32,768)
  __bf16* Yb = (__bf16*)(ws + 30441472);      // 6*26624*512*2 = 163,577,856

  prep_supports<<<dim3(32, 32, 3), dim3(16, 16), 0, stream>>>(s0, s1, s2, At);
  convert_x   <<<dim3(2048),       dim3(256),   0, stream>>>(x, Xb);
  convert_W   <<<dim3(56),         dim3(256),   0, stream>>>(W, Wb);
  prop_gemm   <<<dim3(208, 4, 6),  dim3(256),   0, stream>>>(Xb, At, Yb);
  out_gemm    <<<dim3(64, 64),     dim3(256),   0, stream>>>(x, Yb, Wb, b, out);
}
